// RNNClassifier_86938728006087
// MI455X (gfx1250) — compile-verified, matrix operand order checked by reading, not run on hardware
//
#include <hip/hip_runtime.h>
#include <hip/hip_bf16.h>

typedef __attribute__((ext_vector_type(16))) __bf16 v16bf;
typedef __attribute__((ext_vector_type(8)))  float  v8f;
typedef unsigned __attribute__((ext_vector_type(4))) uint4v;
typedef int      __attribute__((ext_vector_type(4))) int4v;
typedef int      __attribute__((ext_vector_type(8))) int8v;

#define LSTM_B   64
#define LSTM_T   512
#define LSTM_IN  512
#define LSTM_H   1024
#define LSTM_G   4096
#define LSTM_NC  1000
#define REC_NWG  32

// ---------------------------------------------------------------------------
// WMMA helpers (CDNA5 wave32, 16x16x32 bf16, fp32 accumulate)
// ---------------------------------------------------------------------------
__device__ inline v8f wmma_bf16(v16bf a, v16bf b, v8f c) {
  return __builtin_amdgcn_wmma_f32_16x16x32_bf16(
      /*neg_a=*/false, a, /*neg_b=*/false, b,
      /*c_mod=*/(short)0, c, /*reuse_a=*/false, /*reuse_b=*/false);
}

// Load one 16x32 bf16 fragment from a row-major matrix (K contiguous).
// ISA layout (16-bit A 16x32): lane = 16*c + r (r = row 0..15, c = lane group),
// elements 0..7  hold K = k0 + 8c + 0..7   (one 16B chunk)
// elements 8..15 hold K = k0 + 16 + 8c + 0..7 (second 16B chunk)
// B fragments for C = A*B^T use the identical per-row layout (rows of W).
__device__ inline v16bf load_frag_rm(const __bf16* __restrict__ base,
                                     int row0, int ld, int k0, int rowClamp) {
  int lane = threadIdx.x & 31;
  int r = lane & 15;
  int c = lane >> 4;
  int row = row0 + r;
  if (row > rowClamp) row = rowClamp;
  const __bf16* p = base + (size_t)row * ld + k0 + 8 * c;
  union { v16bf v; uint4 q[2]; } u;
  u.q[0] = *(const uint4*)(p);
  u.q[1] = *(const uint4*)(p + 16);
  return u.v;
}

// ---------------------------------------------------------------------------
// Elementwise utility kernels
// ---------------------------------------------------------------------------
__global__ void cvt_bf16_kernel(const float* __restrict__ src,
                                __bf16* __restrict__ dst, size_t n) {
  size_t i = (size_t)blockIdx.x * blockDim.x + threadIdx.x;
  size_t stride = (size_t)gridDim.x * blockDim.x;
  for (; i < n; i += stride) dst[i] = (__bf16)src[i];
}

__global__ void add_bias_kernel(const float* __restrict__ a,
                                const float* __restrict__ b,
                                float* __restrict__ out, int n) {
  int i = blockIdx.x * blockDim.x + threadIdx.x;
  if (i < n) out[i] = a[i] + b[i];
}

__global__ void init_barrier_kernel(unsigned* p) { p[0] = 0u; p[1] = 0u; }

// ---------------------------------------------------------------------------
// Generic bf16 WMMA GEMM:  C[M,N] = A[M,K] * B[N,K]^T + bias[N]
// Block tile 128x128, 256 threads = 8 waves, each wave: 2 m-tiles x 4 n-tiles.
// ---------------------------------------------------------------------------
__global__ __launch_bounds__(256)
void gemm_bf16_kernel(const __bf16* __restrict__ A, const __bf16* __restrict__ B,
                      const float* __restrict__ bias, float* __restrict__ C,
                      int M, int N, int K, int lda, int ldb, int ldc) {
  int wave = threadIdx.x >> 5;
  int lane = threadIdx.x & 31;
  int bm = blockIdx.y * 128;
  int bn = blockIdx.x * 128;
  int mt0 = (wave & 3) * 2;   // two consecutive m-tiles
  int nt0 = (wave >> 2) * 4;  // four consecutive n-tiles

  v8f acc[2][4];
#pragma unroll
  for (int i = 0; i < 2; i++)
#pragma unroll
    for (int j = 0; j < 4; j++) acc[i][j] = (v8f){};

  for (int k0 = 0; k0 < K; k0 += 32) {
    v16bf a0 = load_frag_rm(A, bm + (mt0 + 0) * 16, lda, k0, M - 1);
    v16bf a1 = load_frag_rm(A, bm + (mt0 + 1) * 16, lda, k0, M - 1);
#pragma unroll
    for (int j = 0; j < 4; j++) {
      v16bf bf = load_frag_rm(B, bn + (nt0 + j) * 16, ldb, k0, N - 1);
      acc[0][j] = wmma_bf16(a0, bf, acc[0][j]);
      acc[1][j] = wmma_bf16(a1, bf, acc[1][j]);
    }
  }

  int hi = lane >> 4;
  int nn = lane & 15;
#pragma unroll
  for (int i = 0; i < 2; i++)
#pragma unroll
    for (int j = 0; j < 4; j++) {
      int n = bn + (nt0 + j) * 16 + nn;
      if (n >= N) continue;
      float bv = bias ? bias[n] : 0.f;
      int mb = bm + (mt0 + i) * 16 + 8 * hi;
#pragma unroll
      for (int e = 0; e < 8; e++) {
        int m = mb + e;
        if (m < M) C[(size_t)m * ldc + n] = acc[i][j][e] + bv;
      }
    }
}

// ---------------------------------------------------------------------------
// Software grid barrier (32 persistent workgroups, agent-scope atomics)
// ---------------------------------------------------------------------------
__device__ inline void grid_barrier(unsigned* cnt, unsigned* sense, unsigned nwg) {
  __syncthreads();
  if (threadIdx.x == 0) {
    __threadfence();  // make h stores visible device-wide
    unsigned gen = __hip_atomic_load(sense, __ATOMIC_ACQUIRE, __HIP_MEMORY_SCOPE_AGENT);
    unsigned arr = __hip_atomic_fetch_add(cnt, 1u, __ATOMIC_ACQ_REL, __HIP_MEMORY_SCOPE_AGENT);
    if (arr == nwg - 1) {
      __hip_atomic_store(cnt, 0u, __ATOMIC_RELAXED, __HIP_MEMORY_SCOPE_AGENT);
      __hip_atomic_fetch_add(sense, 1u, __ATOMIC_ACQ_REL, __HIP_MEMORY_SCOPE_AGENT);
    } else {
      while (__hip_atomic_load(sense, __ATOMIC_ACQUIRE, __HIP_MEMORY_SCOPE_AGENT) == gen)
        __builtin_amdgcn_s_sleep(1);
    }
  }
  __syncthreads();
}

// ---------------------------------------------------------------------------
// Persistent LSTM recurrence. 32 WGs x 256 threads; WG wg owns hidden units
// [wg*32, wg*32+32). Its 128 gate rows of W_hh (128x1024 bf16 = 256 KB) stay
// resident in LDS for all 512 steps (CDNA5: 320 KB LDS per WGP). The slice is
// staged by the Tensor Data Mover (one 2D D# per gate block) when available.
// Per step: gates = XG[:,t] + h @ Whh_slice^T via WMMA; fused cell update.
// ---------------------------------------------------------------------------
__global__ __launch_bounds__(256)
void lstm_rec_kernel(const float* __restrict__ XG,      // [B*T, 4H], row = b*T+t
                     const __bf16* __restrict__ Whh,    // [4H, H] bf16 row-major
                     __bf16* __restrict__ hbuf,         // [B, H] bf16
                     __bf16* __restrict__ outbuf,       // [B, T, H] bf16
                     int storeOut,
                     unsigned* __restrict__ barCnt, unsigned* __restrict__ barSense) {
  extern __shared__ __bf16 smem[];                 // [128][1024] weights (256 KB)
  float* Gs = (float*)(smem + 128 * LSTM_H);       // [64][128] gate staging (32 KB)

  const int wg = blockIdx.x;
  const int n0 = wg * 32;
  const int tid = threadIdx.x;
  const int wave = tid >> 5;
  const int lane = tid & 31;

  // --- preload weight slice into LDS: local row r -> global row (r>>5)*H + n0 + (r&31)
#if __has_builtin(__builtin_amdgcn_tensor_load_to_lds) && __has_builtin(__builtin_amdgcn_s_wait_tensorcnt)
  if (wave == 0) {  // TDM ignores EXEC; issue the 4 DMA descriptors from one wave
    // byte offset of our dynamic-LDS allocation within the wave's LDS space
    typedef __attribute__((address_space(3))) char lds_char;
    unsigned ldsBase = (unsigned)(unsigned long long)(lds_char*)smem;
#pragma unroll
    for (int g = 0; g < 4; ++g) {
      unsigned long long ga =
          (unsigned long long)(const void*)(Whh + (size_t)(g * LSTM_H + n0) * LSTM_H);
      // D# group 0: count=1 | lds_addr | global_addr | type=2 ("image")
      uint4v g0;
      g0[0] = 1u;                                        // count=1, user mode
      g0[1] = ldsBase + (unsigned)g * 32u * LSTM_H * 2u; // LDS byte address
      g0[2] = (unsigned)(ga & 0xffffffffu);              // global_addr[31:0]
      g0[3] = (unsigned)((ga >> 32) & 0x01ffffffu) | (2u << 30);  // addr[56:32]|type
      // D# group 1: data_size=2B; tensor 1024 x 32, stride 1024; tile 1024 x 32
      int8v g1;
      g1[0] = (int)(1u << 16);           // wg_mask=0, data_size=1 (2 bytes)
      g1[1] = (int)(1024u << 16);        // tensor_dim0 = 1024 (low bits at 48)
      g1[2] = (int)(32u << 16);          // tensor_dim0 hi=0, tensor_dim1 = 32
      g1[3] = (int)(1024u << 16);        // tensor_dim1 hi=0, tile_dim0 = 1024
      g1[4] = (int)32u;                  // tile_dim1 = 32, tile_dim2 = 0
      g1[5] = (int)1024u;                // tensor_dim0_stride = 1024
      g1[6] = 0;                         // stride hi, tensor_dim1_stride lo
      g1[7] = 0;
      int4v gz4 = (int4v){0, 0, 0, 0};
      int8v gz8 = (int8v){0, 0, 0, 0, 0, 0, 0, 0};
      // 6-arg form (clang-23 / therock): (g0, g1, g2, g3, g4, cpol)
      __builtin_amdgcn_tensor_load_to_lds(g0, g1, gz4, gz4, gz8, 0);
    }
  }
#else
  {
    const int q_per_row = LSTM_H * 2 / 16;  // 128 uint4 per row
#pragma unroll 4
    for (int it = 0; it < 64; ++it) {       // 16384 uint4 / 256 threads
      int q = it * 256 + tid;
      int r = q / q_per_row;
      int cq = q % q_per_row;
      int grow = (r >> 5) * LSTM_H + n0 + (r & 31);
      ((uint4*)smem)[q] = ((const uint4*)(Whh + (size_t)grow * LSTM_H))[cq];
    }
  }
#endif

  // --- h0 = 0 for our slice (overlaps with the weight DMA)
  for (int e = tid; e < LSTM_B * 32; e += 256) {
    int b = e >> 5, j = e & 31;
    hbuf[(size_t)b * LSTM_H + n0 + j] = (__bf16)0.f;
  }

#if __has_builtin(__builtin_amdgcn_tensor_load_to_lds) && __has_builtin(__builtin_amdgcn_s_wait_tensorcnt)
  if (wave == 0) __builtin_amdgcn_s_wait_tensorcnt(0);
#endif
  __syncthreads();
  grid_barrier(barCnt, barSense, REC_NWG);

  float c_reg[8];
#pragma unroll
  for (int s = 0; s < 8; s++) c_reg[s] = 0.f;

  const int mt  = wave & 3;         // m-tile (batch rows mt*16..+15)
  const int ntb = (wave >> 2) * 4;  // four n-tiles of the 128-wide gate slice
  const int hi = lane >> 4;
  const int nn = lane & 15;

  for (int t = 0; t < LSTM_T; ++t) {
    // --- seed accumulators with XG (input projection + biases)
    v8f acc[4];
#pragma unroll
    for (int j = 0; j < 4; j++) {
      int ncol = (ntb + j) * 16 + nn;                       // 0..127 within slice
      int gcol = (ncol >> 5) * LSTM_H + n0 + (ncol & 31);   // column in [0,4096)
      v8f a = (v8f){};
#pragma unroll
      for (int e = 0; e < 8; e++) {
        int m = mt * 16 + 8 * hi + e;                       // batch index
        a[e] = XG[((size_t)m * LSTM_T + t) * LSTM_G + gcol];
      }
      acc[j] = a;
    }
    // --- speculative prefetch of next step's XG seeds (global_prefetch_b8)
    if (t + 1 < LSTM_T) {
#pragma unroll
      for (int j = 0; j < 4; j++) {
        int ncol = (ntb + j) * 16 + nn;
        int gcol = (ncol >> 5) * LSTM_H + n0 + (ncol & 31);
#pragma unroll
        for (int e = 0; e < 8; e++) {
          int m = mt * 16 + 8 * hi + e;
          __builtin_prefetch(&XG[((size_t)m * LSTM_T + (t + 1)) * LSTM_G + gcol], 0, 0);
        }
      }
    }
    // --- gates += h @ Whh_slice^T  (A from global h, B from LDS weights)
    for (int k0 = 0; k0 < LSTM_H; k0 += 32) {
      v16bf af = load_frag_rm(hbuf, mt * 16, LSTM_H, k0, LSTM_B - 1);
#pragma unroll
      for (int j = 0; j < 4; j++) {
        v16bf bf = load_frag_rm(smem, (ntb + j) * 16, LSTM_H, k0, 127);
        acc[j] = wmma_bf16(af, bf, acc[j]);
      }
    }
    // --- stage gates in LDS so each thread sees all four gates of a unit
#pragma unroll
    for (int j = 0; j < 4; j++) {
      int ncol = (ntb + j) * 16 + nn;
#pragma unroll
      for (int e = 0; e < 8; e++) {
        int m = mt * 16 + 8 * hi + e;
        Gs[m * 128 + ncol] = acc[j][e];
      }
    }
    __syncthreads();
    // --- fused LSTM cell update; c stays in registers for all 512 steps
#pragma unroll
    for (int s = 0; s < 8; s++) {
      int e = tid + s * 256;        // 2048 = 64 batch x 32 units
      int b = e >> 5, j = e & 31;
      float gi = Gs[b * 128 + j];
      float gf = Gs[b * 128 + 32 + j];
      float gg = Gs[b * 128 + 64 + j];
      float go = Gs[b * 128 + 96 + j];
      gi = 1.f / (1.f + __expf(-gi));
      gf = 1.f / (1.f + __expf(-gf));
      gg = tanhf(gg);
      go = 1.f / (1.f + __expf(-go));
      float c = gf * c_reg[s] + gi * gg;
      c_reg[s] = c;
      float h = go * tanhf(c);
      __bf16 hb = (__bf16)h;
      hbuf[(size_t)b * LSTM_H + n0 + j] = hb;
      if (storeOut)
        outbuf[((size_t)b * LSTM_T + t) * LSTM_H + n0 + j] = hb;
    }
    grid_barrier(barCnt, barSense, REC_NWG);  // h visible to all WGs before t+1
  }
}

// ---------------------------------------------------------------------------
// Host side
// ---------------------------------------------------------------------------
extern "C" void kernel_launch(void* const* d_in, const int* in_sizes, int n_in,
                              void* d_out, int out_size, void* d_ws, size_t ws_size,
                              hipStream_t stream) {
  (void)in_sizes; (void)n_in; (void)out_size; (void)ws_size;
  const float* x     = (const float*)d_in[0];
  const float* W_ih0 = (const float*)d_in[1];
  const float* W_hh0 = (const float*)d_in[2];
  const float* b_ih0 = (const float*)d_in[3];
  const float* b_hh0 = (const float*)d_in[4];
  const float* W_ih1 = (const float*)d_in[5];
  const float* W_hh1 = (const float*)d_in[6];
  const float* b_ih1 = (const float*)d_in[7];
  const float* b_hh1 = (const float*)d_in[8];
  const float* fc_w  = (const float*)d_in[9];
  const float* fc_b  = (const float*)d_in[10];
  float* out = (float*)d_out;

  // workspace carve-up (256B aligned)
  char* ws = (char*)d_ws;
  size_t off = 0;
  auto carve = [&](size_t bytes) { size_t o = off; off = (off + bytes + 255) & ~(size_t)255; return o; };
  const size_t MT = (size_t)LSTM_B * LSTM_T;  // 32768
  float*  XG    = (float*)(ws + carve(MT * LSTM_G * 4));          // 512 MB
  __bf16* xbf   = (__bf16*)(ws + carve(MT * LSTM_IN * 2));
  __bf16* out0  = (__bf16*)(ws + carve(MT * LSTM_H * 2));
  __bf16* wih0b = (__bf16*)(ws + carve((size_t)LSTM_G * LSTM_IN * 2));
  __bf16* whh0b = (__bf16*)(ws + carve((size_t)LSTM_G * LSTM_H * 2));
  __bf16* wih1b = (__bf16*)(ws + carve((size_t)LSTM_G * LSTM_H * 2));
  __bf16* whh1b = (__bf16*)(ws + carve((size_t)LSTM_G * LSTM_H * 2));
  __bf16* fcwb  = (__bf16*)(ws + carve((size_t)LSTM_NC * LSTM_H * 2));
  __bf16* hbuf  = (__bf16*)(ws + carve((size_t)LSTM_B * LSTM_H * 2));
  float*  bias0 = (float*)(ws + carve(LSTM_G * 4));
  float*  bias1 = (float*)(ws + carve(LSTM_G * 4));
  unsigned* bar = (unsigned*)(ws + carve(256));

  // 1) bf16 conversions
  cvt_bf16_kernel<<<4096, 256, 0, stream>>>(x,     xbf,   MT * LSTM_IN);
  cvt_bf16_kernel<<<2048, 256, 0, stream>>>(W_ih0, wih0b, (size_t)LSTM_G * LSTM_IN);
  cvt_bf16_kernel<<<4096, 256, 0, stream>>>(W_hh0, whh0b, (size_t)LSTM_G * LSTM_H);
  cvt_bf16_kernel<<<4096, 256, 0, stream>>>(W_ih1, wih1b, (size_t)LSTM_G * LSTM_H);
  cvt_bf16_kernel<<<4096, 256, 0, stream>>>(W_hh1, whh1b, (size_t)LSTM_G * LSTM_H);
  cvt_bf16_kernel<<<1024, 256, 0, stream>>>(fc_w,  fcwb,  (size_t)LSTM_NC * LSTM_H);
  add_bias_kernel<<<16, 256, 0, stream>>>(b_ih0, b_hh0, bias0, LSTM_G);
  add_bias_kernel<<<16, 256, 0, stream>>>(b_ih1, b_hh1, bias1, LSTM_G);

  const size_t recLds = 128 * LSTM_H * 2 + LSTM_B * 128 * 4;  // 294912 B < 320 KB

  // 2) layer 0: input projection for all t, then persistent recurrence
  gemm_bf16_kernel<<<dim3(LSTM_G / 128, MT / 128), 256, 0, stream>>>(
      xbf, wih0b, bias0, XG, (int)MT, LSTM_G, LSTM_IN, LSTM_IN, LSTM_IN, LSTM_G);
  init_barrier_kernel<<<1, 1, 0, stream>>>(bar);
  lstm_rec_kernel<<<REC_NWG, 256, recLds, stream>>>(
      XG, whh0b, hbuf, out0, /*storeOut=*/1, bar, bar + 1);

  // 3) layer 1
  gemm_bf16_kernel<<<dim3(LSTM_G / 128, MT / 128), 256, 0, stream>>>(
      out0, wih1b, bias1, XG, (int)MT, LSTM_G, LSTM_H, LSTM_H, LSTM_H, LSTM_G);
  init_barrier_kernel<<<1, 1, 0, stream>>>(bar);
  lstm_rec_kernel<<<REC_NWG, 256, recLds, stream>>>(
      XG, whh1b, hbuf, out0, /*storeOut=*/0, bar, bar + 1);

  // 4) classifier on final hidden state
  gemm_bf16_kernel<<<dim3((LSTM_NC + 127) / 128, 1), 256, 0, stream>>>(
      hbuf, fcwb, fc_b, out, LSTM_B, LSTM_NC, LSTM_H, LSTM_H, LSTM_H, LSTM_NC);
}
